// QuantFinanceAttention_65000035058567
// MI455X (gfx1250) — matrix operand order repaired and checked by reading, not compiled, not run on hardware
//
#include <hip/hip_runtime.h>
#include <hip/hip_bf16.h>
#include <math.h>

// ---------------- CDNA5 types ----------------
typedef __attribute__((ext_vector_type(16))) __bf16 v16bf;
typedef __attribute__((ext_vector_type(8)))  __bf16 bf16x8;
typedef __attribute__((ext_vector_type(4)))  __bf16 bf16x4;
typedef __attribute__((ext_vector_type(8)))  float  v8f;
typedef __attribute__((ext_vector_type(4)))  float  f32x4;

#define B_   2
#define T_   2048
#define C_   1024
#define H_   16
#define DK_  64
#define SCALE_ 0.125f   // 1/sqrt(64)

static __device__ __forceinline__ v16bf make_frag(bf16x8 lo, bf16x8 hi) {
  union { v16bf v; bf16x8 h[2]; } u;
  u.h[0] = lo; u.h[1] = hi;
  return u.v;
}
static __device__ __forceinline__ v8f wmma_bf16(v16bf a, v16bf b, v8f c) {
  // (neg_a, A, neg_b, B, c_mod, C, reuse_a, reuse_b)
  return __builtin_amdgcn_wmma_f32_16x16x32_bf16(false, a, false, b, (short)0, c,
                                                 false, false);
}

// ---------------- fp32 -> bf16 convert ----------------
__global__ void cvt_f32_bf16(const float* __restrict__ src,
                             __bf16* __restrict__ dst, int n) {
  int i = (blockIdx.x * blockDim.x + threadIdx.x) * 4;
  if (i + 3 < n) {
    f32x4 v = *(const f32x4*)(src + i);
    bf16x4 o;
    o.x = (__bf16)v.x; o.y = (__bf16)v.y; o.z = (__bf16)v.z; o.w = (__bf16)v.w;
    *(bf16x4*)(dst + i) = o;
  } else {
    for (; i < n; ++i) dst[i] = (__bf16)src[i];
  }
}

// ---------------- GEMM: Y = A(bf16, MxK) @ W(bf16, NxK)^T ----------------
// One wave computes a 16x64 tile. The K-loop is ping-pong software-pipelined
// (manual unroll by 2) so no cross-iteration register copies are generated:
// while set P's 4 wmmas issue, set Q's 10 b128 loads are in flight.
// Modes: 0=Q(+RoPE), 1=K(+RoPE), 2=V(transposed store), 3=out-proj(+bias,f32)
__global__ void __launch_bounds__(256, 1)
gemm16x64(const __bf16* __restrict__ A,
          const __bf16* __restrict__ W0, const __bf16* __restrict__ W1,
          const __bf16* __restrict__ W2,
          const float* __restrict__ bias,
          __bf16* __restrict__ qout, __bf16* __restrict__ kout,
          __bf16* __restrict__ vTout, float* __restrict__ fout,
          int mode_base) {
  const int mode = mode_base + (int)blockIdx.y;
  const __bf16* W = (mode == 1) ? W1 : (mode == 2) ? W2 : W0;

  const int lane = threadIdx.x & 31;
  const int wid  = threadIdx.x >> 5;
  const int g    = blockIdx.x * 8 + wid;     // 0..4095 wave tiles
  const int m0   = (g >> 4) * 16;            // row tile (token rows)
  const int n0   = (g & 15) * 64;            // 64-wide column strip (one head)
  const int lm   = lane & 15;
  const int lh   = lane >> 4;

  v8f acc[4] = {};

  const __bf16* arow = A + (size_t)(m0 + lm) * C_;
  const __bf16* wr[4];
#pragma unroll
  for (int t = 0; t < 4; ++t) wr[t] = W + (size_t)(n0 + t * 16 + lm) * C_;

  v16bf aP, bP[4], aQ, bQ[4];

#define LOADSET(av, bv, kk)                                                   \
  do {                                                                        \
    av = make_frag(*(const bf16x8*)(arow + (kk) + 16 * lh),                   \
                   *(const bf16x8*)(arow + (kk) + 16 * lh + 8));              \
    _Pragma("unroll")                                                         \
    for (int t = 0; t < 4; ++t)                                               \
      bv[t] = make_frag(*(const bf16x8*)(wr[t] + (kk) + 16 * lh),             \
                        *(const bf16x8*)(wr[t] + (kk) + 16 * lh + 8));        \
  } while (0)

#define MMASET(av, bv)                                                        \
  do {                                                                        \
    _Pragma("unroll")                                                         \
    for (int t = 0; t < 4; ++t) acc[t] = wmma_bf16(av, bv[t], acc[t]);        \
  } while (0)

  LOADSET(aP, bP, 0);
  int k0 = 32;
  for (; k0 + 32 < C_; k0 += 64) {   // ping-pong: 2 K-steps per iteration
    LOADSET(aQ, bQ, k0);
    MMASET(aP, bP);
    LOADSET(aP, bP, k0 + 32);
    MMASET(aQ, bQ);
  }
  LOADSET(aQ, bQ, k0);               // last K-step (k0 == C_-32)
  MMASET(aP, bP);
  MMASET(aQ, bQ);
#undef LOADSET
#undef MMASET

  // C/D layout: lane holds col n = lm; VGPR j holds row m = j + 8*lh
  const int head = n0 / DK_;
  if (mode <= 1) {  // RoPE + store bf16 [B,H,T,DK]
    __bf16* out = (mode == 0) ? qout : kout;
    // inverse frequencies depend only on lane (d = t*16+lm), hoist out of j
    const float inv0 = __powf(10000.f, -((float)lm) / 32.f);
    const float inv1 = __powf(10000.f, -((float)(16 + lm)) / 32.f);
#pragma unroll
    for (int j = 0; j < 8; ++j) {
      const int gm = m0 + j + 8 * lh;
      const int bb = gm / T_;
      const int tpos = gm % T_;
      __bf16* ob = out + (size_t)((bb * H_ + head) * T_ + tpos) * DK_;
      const float tf = (float)tpos;
#pragma unroll
      for (int t = 0; t < 2; ++t) {
        const int d = t * 16 + lm;                     // 0..31
        float s, c;
        __sincosf(tf * ((t == 0) ? inv0 : inv1), &s, &c);
        const float xlo = acc[t][j];
        const float xhi = acc[t + 2][j];
        ob[d]      = (__bf16)(xlo * c - xhi * s);
        ob[d + 32] = (__bf16)(xhi * c + xlo * s);
      }
    }
  } else if (mode == 2) {  // V transposed [B,H,DK,T]
#pragma unroll
    for (int j = 0; j < 8; ++j) {
      const int gm = m0 + j + 8 * lh;
      const int bb = gm / T_;
      const int tpos = gm % T_;
      __bf16* ob = vTout + (size_t)(bb * H_ + head) * DK_ * T_;
#pragma unroll
      for (int t = 0; t < 4; ++t)
        ob[(size_t)(t * 16 + lm) * T_ + tpos] = (__bf16)acc[t][j];
    }
  } else {  // mode 3: + bias, f32 out [B*T, C]
#pragma unroll
    for (int j = 0; j < 8; ++j) {
      const int gm = m0 + j + 8 * lh;
#pragma unroll
      for (int t = 0; t < 4; ++t) {
        const int n = n0 + t * 16 + lm;
        fout[(size_t)gm * C_ + n] = acc[t][j] + bias[n];
      }
    }
  }
}

// ---------------- Flash attention ----------------
// One wave per 16-row q tile; 32-key chunks. V chunks are staged through LDS
// with double-buffered async copies (GLOBAL_LOAD_ASYNC_TO_LDS_B128 /
// s_wait_asynccnt) so the async engine overlaps the score WMMAs + softmax.
// Per-wave LDS: 2 x 4KB V buffers + 1KB P slab = 9KB; 4 waves/block = 36KB.
#define WAVE_LDS 9216
__global__ void __launch_bounds__(128, 1)
flash_attn(const __bf16* __restrict__ q, const __bf16* __restrict__ k,
           const __bf16* __restrict__ vT, const float* __restrict__ sbias,
           const float* __restrict__ sw, __bf16* __restrict__ aout) {
  __shared__ __align__(16) unsigned char smem[4 * WAVE_LDS];
  const int lane = threadIdx.x & 31;
  const int wid  = threadIdx.x >> 5;          // 0..3
  const int g    = blockIdx.x * 4 + wid;      // 0..4095
  const int bb   = g >> 11;
  const int h    = (g >> 7) & 15;
  const int q0   = (g & 127) * 16;
  const int lm   = lane & 15;
  const int lh   = lane >> 4;

  const __bf16* qp = q  + (size_t)(bb * H_ + h) * T_ * DK_;
  const __bf16* kp = k  + (size_t)(bb * H_ + h) * T_ * DK_;
  const __bf16* vp = vT + (size_t)(bb * H_ + h) * DK_ * T_;
  const float bias = sbias[h];
  const float* wrow = sw + bb * T_;

  unsigned char* wbase = &smem[wid * WAVE_LDS];
  const unsigned vbase = (unsigned)(size_t)wbase;      // LDS byte address
  __bf16* plds = (__bf16*)(wbase + 8192);              // 16x32 P slab

  // Q A-fragments for the two 32-wide halves of d_k (loop-invariant)
  const __bf16* qrow = qp + (size_t)(q0 + lm) * DK_;
  const v16bf aq0 = make_frag(*(const bf16x8*)(qrow + 16 * lh),
                              *(const bf16x8*)(qrow + 16 * lh + 8));
  const v16bf aq1 = make_frag(*(const bf16x8*)(qrow + 32 + 16 * lh),
                              *(const bf16x8*)(qrow + 32 + 16 * lh + 8));

  v8f o[4] = {};
  float rmax[8], rsum[8];
#pragma unroll
  for (int j = 0; j < 8; ++j) { rmax[j] = -1e30f; rsum[j] = 0.f; }

  const int nch = (q0 + 16 + 31) >> 5;   // number of 32-key chunks (causal)

  // async-stage V chunk `ic` into buffer `buf` (8 x B128, lane-preserving)
  auto issue_v = [&](int k0, int buf) {
#pragma unroll
    for (int t = 0; t < 4; ++t)
#pragma unroll
      for (int p = 0; p < 2; ++p) {
        const __bf16* gsrc = vp + (size_t)(t * 16 + lm) * T_ + k0 + 16 * lh + 8 * p;
        const unsigned ldst = vbase + (unsigned)(buf * 4096)
                            + (unsigned)(((t * 2 + p) * 32 + lane) * 16);
        asm volatile("global_load_async_to_lds_b128 %0, %1, off"
                     :: "v"(ldst), "v"(gsrc) : "memory");
      }
  };

  issue_v(0, 0);

  for (int ic = 0; ic < nch; ++ic) {
    const int k0 = ic * 32;
    const bool more = (ic + 1) < nch;
    if (more) issue_v(k0 + 32, (ic + 1) & 1);

    // ---- scores S = q @ k^T, 16x32 chunk (direct global fragment loads) ----
    v8f s0 = {}, s1 = {};
    {
      const __bf16* kr0 = kp + (size_t)(k0 + lm) * DK_;
      v16bf b00 = make_frag(*(const bf16x8*)(kr0 + 16 * lh),
                            *(const bf16x8*)(kr0 + 16 * lh + 8));
      v16bf b01 = make_frag(*(const bf16x8*)(kr0 + 32 + 16 * lh),
                            *(const bf16x8*)(kr0 + 32 + 16 * lh + 8));
      const __bf16* kr1 = kr0 + 16 * DK_;
      v16bf b10 = make_frag(*(const bf16x8*)(kr1 + 16 * lh),
                            *(const bf16x8*)(kr1 + 16 * lh + 8));
      v16bf b11 = make_frag(*(const bf16x8*)(kr1 + 32 + 16 * lh),
                            *(const bf16x8*)(kr1 + 32 + 16 * lh + 8));
      s0 = wmma_bf16(aq0, b00, s0);
      s0 = wmma_bf16(aq1, b01, s0);
      s1 = wmma_bf16(aq0, b10, s1);
      s1 = wmma_bf16(aq1, b11, s1);
    }

    const int key0 = k0 + lm, key1 = k0 + 16 + lm;
    const float w0 = wrow[key0], w1 = wrow[key1];
    float p0[8], p1[8];
#pragma unroll
    for (int j = 0; j < 8; ++j) {
      const int qg = q0 + j + 8 * lh;
      float e0 = (key0 <= qg) ? (s0[j] * SCALE_ + bias) * w0 : -1e30f;
      float e1 = (key1 <= qg) ? (s1[j] * SCALE_ + bias) * w1 : -1e30f;
      float cm = fmaxf(e0, e1);   // row max over 16 lanes (intra-half masks)
      cm = fmaxf(cm, __shfl_xor(cm, 1, 32));
      cm = fmaxf(cm, __shfl_xor(cm, 2, 32));
      cm = fmaxf(cm, __shfl_xor(cm, 4, 32));
      cm = fmaxf(cm, __shfl_xor(cm, 8, 32));
      const float nm = fmaxf(rmax[j], cm);
      const float alpha = __expf(rmax[j] - nm);
      rmax[j] = nm;
      const float pe0 = __expf(e0 - nm);
      const float pe1 = __expf(e1 - nm);
      float ps = pe0 + pe1;
      ps += __shfl_xor(ps, 1, 32);
      ps += __shfl_xor(ps, 2, 32);
      ps += __shfl_xor(ps, 4, 32);
      ps += __shfl_xor(ps, 8, 32);
      rsum[j] = rsum[j] * alpha + ps;
      p0[j] = pe0; p1[j] = pe1;
#pragma unroll
      for (int t = 0; t < 4; ++t) o[t][j] *= alpha;
    }

    // ---- P: C-layout -> A-layout transpose through LDS ----
#pragma unroll
    for (int j = 0; j < 8; ++j) {
      const int m = j + 8 * lh;
      plds[m * 32 + lm]      = (__bf16)p0[j];
      plds[m * 32 + 16 + lm] = (__bf16)p1[j];
    }
    asm volatile("s_wait_dscnt 0x0" ::: "memory");
    const v16bf pa = make_frag(*(const bf16x8*)(plds + lm * 32 + 16 * lh),
                               *(const bf16x8*)(plds + lm * 32 + 16 * lh + 8));

    // ---- wait for this chunk's async V (leave next chunk's 8 in flight) ----
    if (more) asm volatile("s_wait_asynccnt 0x8" ::: "memory");
    else      asm volatile("s_wait_asynccnt 0x0" ::: "memory");

    // ---- O += P @ V from LDS staging buffer ----
    const unsigned char* vbuf = wbase + (ic & 1) * 4096;
#pragma unroll
    for (int t = 0; t < 4; ++t) {
      bf16x8 lo = *(const bf16x8*)(vbuf + ((t * 2 + 0) * 32 + lane) * 16);
      bf16x8 hi = *(const bf16x8*)(vbuf + ((t * 2 + 1) * 32 + lane) * 16);
      o[t] = wmma_bf16(pa, make_frag(lo, hi), o[t]);
    }
    asm volatile("s_wait_dscnt 0x0" ::: "memory");  // P/V reads done before reuse
  }

  // ---- normalize and store bf16 [B,T,C] ----
#pragma unroll
  for (int j = 0; j < 8; ++j) {
    const int gm = q0 + j + 8 * lh;
    const float inv = 1.f / rsum[j];
    __bf16* ob = aout + (size_t)(bb * T_ + gm) * C_ + h * DK_;
#pragma unroll
    for (int t = 0; t < 4; ++t) ob[t * 16 + lm] = (__bf16)(o[t][j] * inv);
  }
}

// ---------------- host launch ----------------
extern "C" void kernel_launch(void* const* d_in, const int* in_sizes, int n_in,
                              void* d_out, int out_size, void* d_ws, size_t ws_size,
                              hipStream_t stream) {
  const float* x  = (const float*)d_in[0];
  const float* Wq = (const float*)d_in[1];
  const float* Wk = (const float*)d_in[2];
  const float* Wv = (const float*)d_in[3];
  const float* Wo = (const float*)d_in[4];
  const float* bo = (const float*)d_in[5];
  const float* sb = (const float*)d_in[6];   // sector_bias [H]
  const float* sw = (const float*)d_in[7];   // sector_weights [B,T]
  // d_in[8]: causal mask — implied by construction, not read.

  char* ws = (char*)d_ws;                    // needs 48 MB
  __bf16* xb  = (__bf16*)(ws);               //  8 MB  x bf16
  __bf16* Wqb = (__bf16*)(ws + (8u  << 20)); //  2 MB
  __bf16* Wkb = (__bf16*)(ws + (10u << 20)); //  2 MB
  __bf16* Wvb = (__bf16*)(ws + (12u << 20)); //  2 MB
  __bf16* Wob = (__bf16*)(ws + (14u << 20)); //  2 MB
  __bf16* qb  = (__bf16*)(ws + (16u << 20)); //  8 MB  [B,H,T,DK]
  __bf16* kb  = (__bf16*)(ws + (24u << 20)); //  8 MB  [B,H,T,DK]
  __bf16* vTb = (__bf16*)(ws + (32u << 20)); //  8 MB  [B,H,DK,T]
  __bf16* ab  = (__bf16*)(ws + (40u << 20)); //  8 MB  attn out [B,T,C]

  const int nx = B_ * T_ * C_;               // 4 Mi
  const int nw = C_ * C_;                    // 1 Mi
  cvt_f32_bf16<<<nx / 1024, 256, 0, stream>>>(x,  xb,  nx);
  cvt_f32_bf16<<<nw / 1024, 256, 0, stream>>>(Wq, Wqb, nw);
  cvt_f32_bf16<<<nw / 1024, 256, 0, stream>>>(Wk, Wkb, nw);
  cvt_f32_bf16<<<nw / 1024, 256, 0, stream>>>(Wv, Wvb, nw);
  cvt_f32_bf16<<<nw / 1024, 256, 0, stream>>>(Wo, Wob, nw);

  // Q/K/V projections (+RoPE, V transposed): 4096 wave-tiles x 3 matrices
  gemm16x64<<<dim3(512, 3), 256, 0, stream>>>(xb, Wqb, Wkb, Wvb, bo,
                                              qb, kb, vTb, (float*)d_out, 0);

  // Flash attention: one wave per 16-row q tile, 4 waves/block
  flash_attn<<<1024, 128, 0, stream>>>(qb, kb, vTb, sb, sw, ab);

  // Output projection + bias -> f32 d_out
  gemm16x64<<<dim3(512, 1), 256, 0, stream>>>(ab, Wob, Wob, Wob, bo,
                                              qb, kb, vTb, (float*)d_out, 3);
}
